// DistanceWeightedGNN_6090263625952
// MI455X (gfx1250) — matrix operand order repaired and checked by reading, not compiled, not run on hardware
//
#include <hip/hip_runtime.h>
#include <hip/hip_bf16.h>

typedef __attribute__((ext_vector_type(16))) _Float16 v16h;
typedef __attribute__((ext_vector_type(8)))  _Float16 v8h;
typedef __attribute__((ext_vector_type(8)))  float    v8f;
typedef __attribute__((ext_vector_type(4)))  float    v4f;
typedef __attribute__((ext_vector_type(4)))  unsigned int u32x4;
typedef __attribute__((ext_vector_type(8)))  int      i32x8;
typedef __attribute__((ext_vector_type(4)))  int      i32x4;

#define IN_C  512
#define HID   64
#define OP_C  64
#define OUT_C 2

#define HAVE_TDM __has_builtin(__builtin_amdgcn_tensor_load_to_lds)

// ---------------------------------------------------------------- TDM -----
// Copy `elems` f16 elements from global `gptr` to LDS `lptr` with the Tensor
// Data Mover, described as a 1-row 2D tensor (dim0 = elems, data_size = 2B).
// D# built per CDNA5 ISA.
// This toolchain's builtin is the 6-arg form:
//   (u32x4 g0, i32x8 g1, i32x4 g2, i32x4 g3, i32x8 extra, i32 cpol)
__device__ __forceinline__ void tdm_copy_f16(const _Float16* gptr, _Float16* lptr,
                                             unsigned elems) {
#if HAVE_TDM
    unsigned lds = (unsigned)(uintptr_t)lptr;                 // generic LDS ptr: low 32b = LDS byte addr
    unsigned long long ga = (unsigned long long)(uintptr_t)gptr;
    u32x4 g0;
    g0[0] = 1u;                                               // count=1 (valid user descriptor)
    g0[1] = lds;                                              // lds_addr [63:32]
    g0[2] = (unsigned)(ga & 0xFFFFFFFFu);                     // global_addr [95:64]
    g0[3] = (unsigned)((ga >> 32) & 0x01FFFFFFu)              // global_addr [120:96]
          | 0x80000000u;                                      // type=2 ("image") [127:126]
    i32x8 g1;
    g1[0] = 0x00010000;                                       // data_size=1 (2 bytes) bits[17:16]
    g1[1] = (int)((elems & 0xFFFFu) << 16);                   // tensor_dim0[15:0] in bits[63:48]
    g1[2] = (int)(((elems >> 16) & 0xFFFFu) | (1u << 16));    // tensor_dim0[31:16] | tensor_dim1=1
    g1[3] = (int)((elems & 0xFFFFu) << 16);                   // tile_dim0 = elems bits[127:112]
    g1[4] = 1;                                                // tile_dim1=1, tile_dim2=0
    g1[5] = (int)elems;                                       // tensor_dim0_stride[31:0]
    g1[6] = 0;                                                // stride hi / dim1_stride lo
    g1[7] = 0;
    i32x4 z4 = {0, 0, 0, 0};
    i32x8 z8 = {0, 0, 0, 0, 0, 0, 0, 0};
    __builtin_amdgcn_tensor_load_to_lds(g0, g1, z4, z4, z8, 0);
    __builtin_amdgcn_s_wait_tensorcnt(0);
#else
    // cooperative fallback: 8 halves (16B) per thread per step
    for (unsigned i = threadIdx.x * 8u; i < elems; i += 256u * 8u)
        *(v8h*)(lptr + i) = *(const v8h*)(gptr + i);
#endif
}

// ---------------------------------------------------------------- prep ----
__global__ void k_init_deg(float* deg, int n) {
    int i = blockIdx.x * 256 + threadIdx.x;
    if (i < n) deg[i] = 1.0f;              // self-loop weight
}

__global__ void k_edge_w_deg(const float* __restrict__ ea, const int* __restrict__ ei,
                             float* __restrict__ w, float* deg, int E) {
    int e = blockIdx.x * 256 + threadIdx.x;
    if (e >= E) return;
    float we = 1.0f / (ea[e] + 1.0f);
    w[e] = we;
    atomicAdd(&deg[ei[E + e]], we);        // col = target
}

__global__ void k_dinv(float* deg, int n) {
    int i = blockIdx.x * 256 + threadIdx.x;
    if (i >= n) return;
    float d = deg[i];
    deg[i] = d > 0.0f ? rsqrtf(fmaxf(d, 1e-30f)) : 0.0f;
}

// W [K x HID] row-major f32  ->  Wt [HID x K] f16 (col-contiguous in K)
__global__ void k_transpose_f16(const float* __restrict__ W, _Float16* __restrict__ Wt,
                                int K, int Ncols) {
    int i = blockIdx.x * 256 + threadIdx.x;
    if (i >= K * Ncols) return;
    int k = i / Ncols, c = i % Ncols;
    Wt[c * K + k] = (_Float16)W[i];
}

// Wcomb[i][j] = sum_k Wp[i][k] * Wfc[k][j]   (k<512);  bias2 = bfc + bp@Wfc_top
__global__ void k_prep_final(const float* __restrict__ Wp, const float* __restrict__ bp,
                             const float* __restrict__ Wfc, const float* __restrict__ bfc,
                             float* Wcomb, float* bias2) {
    int t = threadIdx.x;
    if (t < HID * OUT_C) {
        int i = t >> 1, j = t & 1;
        float s = 0.0f;
        for (int k = 0; k < IN_C; ++k) s += Wp[i * IN_C + k] * Wfc[k * OUT_C + j];
        Wcomb[t] = s;
    } else if (t < HID * OUT_C + OUT_C) {
        int j = t - HID * OUT_C;
        float s = bfc[j];
        for (int k = 0; k < IN_C; ++k) s += bp[k] * Wfc[k * OUT_C + j];
        bias2[j] = s;
    }
}

// ------------------------------------------------------------- GEMM 1 ----
// X[n,512] f32 @ Bt[64][512] f16 -> Out[n,64] f32.
// Block = 256 thr = 8 waves; each wave: 16 rows x 64 cols (1 A frag, 4 accs).
// B staged once per block into LDS via TDM; X read exactly once from HBM.
__global__ void k_gemm_x_w1(const float* __restrict__ X, const _Float16* __restrict__ Bt,
                            float* __restrict__ Out, int n) {
    __shared__ _Float16 Bs[HID * IN_C];                       // 64 KB
#if HAVE_TDM
    if (threadIdx.x < 32) tdm_copy_f16(Bt, Bs, HID * IN_C);
#else
    tdm_copy_f16(Bt, Bs, HID * IN_C);
#endif
    __syncthreads();

    const int lane  = threadIdx.x & 31;
    const int wave  = threadIdx.x >> 5;
    const int mbase = blockIdx.x * 128 + wave * 16;
    if (mbase >= n) return;
    int row = mbase + (lane & 15);
    if (row >= n) row = n - 1;                                // clamp loads; stores predicated
    const int hi = lane >> 4;                                 // K-half select (contiguous k order: k0+16*hi+s)
    const float*    xrow = X + (size_t)row * IN_C;
    const _Float16* b0p  = Bs + (size_t)( 0 + (lane & 15)) * IN_C + hi * 16;
    const _Float16* b1p  = Bs + (size_t)(16 + (lane & 15)) * IN_C + hi * 16;
    const _Float16* b2p  = Bs + (size_t)(32 + (lane & 15)) * IN_C + hi * 16;
    const _Float16* b3p  = Bs + (size_t)(48 + (lane & 15)) * IN_C + hi * 16;

    v8f acc0 = {}, acc1 = {}, acc2 = {}, acc3 = {};
#pragma unroll
    for (int k0 = 0; k0 < IN_C; k0 += 32) {
        // A: lanes 0-15 hold K {k0..k0+15}; lanes 16-31 hold K {k0+16..k0+31}
        v4f f0 = *(const v4f*)(xrow + k0 + hi * 16);
        v4f f1 = *(const v4f*)(xrow + k0 + hi * 16 + 4);
        v4f f2 = *(const v4f*)(xrow + k0 + hi * 16 + 8);
        v4f f3 = *(const v4f*)(xrow + k0 + hi * 16 + 12);
        v16h a;
#pragma unroll
        for (int i = 0; i < 4; ++i) {
            a[i]      = (_Float16)f0[i];
            a[4 + i]  = (_Float16)f1[i];
            a[8 + i]  = (_Float16)f2[i];
            a[12 + i] = (_Float16)f3[i];
        }
        v16h b0 = *(const v16h*)(b0p + k0);                   // ds_load pairs, 32B aligned
        v16h b1 = *(const v16h*)(b1p + k0);
        v16h b2 = *(const v16h*)(b2p + k0);
        v16h b3 = *(const v16h*)(b3p + k0);
        acc0 = __builtin_amdgcn_wmma_f32_16x16x32_f16(false, a, false, b0, (short)0, acc0, false, false);
        acc1 = __builtin_amdgcn_wmma_f32_16x16x32_f16(false, a, false, b1, (short)0, acc1, false, false);
        acc2 = __builtin_amdgcn_wmma_f32_16x16x32_f16(false, a, false, b2, (short)0, acc2, false, false);
        acc3 = __builtin_amdgcn_wmma_f32_16x16x32_f16(false, a, false, b3, (short)0, acc3, false, false);
        if (k0 + 32 < IN_C) __builtin_prefetch(xrow + k0 + 32, 0, 0);
    }
    const int ncol = lane & 15;
    const int orow = mbase + hi * 8;                          // D: VGPR r -> M = r + hi*8
#pragma unroll
    for (int r = 0; r < 8; ++r) {
        int rr = orow + r;
        if (rr < n) {
            float* o = Out + (size_t)rr * HID + ncol;
            o[0]  = acc0[r];
            o[16] = acc1[r];
            o[32] = acc2[r];
            o[48] = acc3[r];
        }
    }
}

// ------------------------------------------------------------- GEMM 2 ----
// H[n,64] f16 @ Bt[64][64] f16 -> Out[n,64] f32. Same structure, K=64.
__global__ void k_gemm_h_w2(const _Float16* __restrict__ H, const _Float16* __restrict__ Bt,
                            float* __restrict__ Out, int n) {
    __shared__ _Float16 Bs[HID * HID];                        // 8 KB
#if HAVE_TDM
    if (threadIdx.x < 32) tdm_copy_f16(Bt, Bs, HID * HID);
#else
    tdm_copy_f16(Bt, Bs, HID * HID);
#endif
    __syncthreads();

    const int lane  = threadIdx.x & 31;
    const int wave  = threadIdx.x >> 5;
    const int mbase = blockIdx.x * 128 + wave * 16;
    if (mbase >= n) return;
    int row = mbase + (lane & 15);
    if (row >= n) row = n - 1;
    const int hi = lane >> 4;
    const _Float16* arow = H + (size_t)row * HID;
    const _Float16* b0p  = Bs + (size_t)( 0 + (lane & 15)) * HID + hi * 16;
    const _Float16* b1p  = Bs + (size_t)(16 + (lane & 15)) * HID + hi * 16;
    const _Float16* b2p  = Bs + (size_t)(32 + (lane & 15)) * HID + hi * 16;
    const _Float16* b3p  = Bs + (size_t)(48 + (lane & 15)) * HID + hi * 16;

    v8f acc0 = {}, acc1 = {}, acc2 = {}, acc3 = {};
#pragma unroll
    for (int k0 = 0; k0 < HID; k0 += 32) {
        v8h a0 = *(const v8h*)(arow + k0 + hi * 16);
        v8h a1 = *(const v8h*)(arow + k0 + hi * 16 + 8);
        v16h a;
#pragma unroll
        for (int i = 0; i < 8; ++i) { a[i] = a0[i]; a[8 + i] = a1[i]; }
        v16h b0 = *(const v16h*)(b0p + k0);
        v16h b1 = *(const v16h*)(b1p + k0);
        v16h b2 = *(const v16h*)(b2p + k0);
        v16h b3 = *(const v16h*)(b3p + k0);
        acc0 = __builtin_amdgcn_wmma_f32_16x16x32_f16(false, a, false, b0, (short)0, acc0, false, false);
        acc1 = __builtin_amdgcn_wmma_f32_16x16x32_f16(false, a, false, b1, (short)0, acc1, false, false);
        acc2 = __builtin_amdgcn_wmma_f32_16x16x32_f16(false, a, false, b2, (short)0, acc2, false, false);
        acc3 = __builtin_amdgcn_wmma_f32_16x16x32_f16(false, a, false, b3, (short)0, acc3, false, false);
    }
    const int ncol = lane & 15;
    const int orow = mbase + hi * 8;
#pragma unroll
    for (int r = 0; r < 8; ++r) {
        int rr = orow + r;
        if (rr < n) {
            float* o = Out + (size_t)rr * HID + ncol;
            o[0]  = acc0[r];
            o[16] = acc1[r];
            o[32] = acc2[r];
            o[48] = acc3[r];
        }
    }
}

// -------------------------------------------------------- aggregation ----
__global__ void k_agg_self(const float* __restrict__ H, const float* __restrict__ dinv,
                           float* __restrict__ Out, int n) {
    int i = blockIdx.x * 256 + threadIdx.x;
    if (i >= n * HID) return;
    float d = dinv[i >> 6];
    Out[i] = d * d * H[i];                     // self-loop: norm = dinv*1*dinv
}

__global__ void k_agg_edges(const float* __restrict__ H, const int* __restrict__ ei,
                            const float* __restrict__ w, const float* __restrict__ dinv,
                            float* __restrict__ Out, int E) {
    int idx = blockIdx.x * 256 + threadIdx.x;
    int e = idx >> 6;
    if (e >= E) return;
    int c  = idx & 63;
    int r  = ei[e];
    int cc = ei[E + e];
    float coef = dinv[r] * w[e] * dinv[cc];
    atomicAdd(&Out[(size_t)cc * HID + c], coef * H[(size_t)r * HID + c]);
}

__global__ void k_bias_relu_f16(const float* __restrict__ A, const float* __restrict__ b,
                                _Float16* __restrict__ H, int n) {
    int i = blockIdx.x * 256 + threadIdx.x;
    if (i >= n * HID) return;
    float v = A[i] + b[i & 63];
    H[i] = (_Float16)(v > 0.0f ? v : 0.0f);
}

__global__ void k_bias_relu_f32(const float* __restrict__ A, const float* __restrict__ b,
                                float* __restrict__ H, int n) {
    int i = blockIdx.x * 256 + threadIdx.x;
    if (i >= n * HID) return;
    float v = A[i] + b[i & 63];
    H[i] = v > 0.0f ? v : 0.0f;
}

// ---------------------------------------------------------------- final ---
__global__ void k_final(const float* __restrict__ H2, const float* __restrict__ op,
                        const float* __restrict__ Wcomb, const float* __restrict__ Wfc,
                        const float* __restrict__ bias2, float* __restrict__ out, int n) {
    int i = blockIdx.x * 256 + threadIdx.x;
    if (i >= n) return;
    float a0 = bias2[0], a1 = bias2[1];
    const float* h = H2 + (size_t)i * HID;
    const float* o = op + (size_t)i * OP_C;
#pragma unroll 8
    for (int c = 0; c < HID; ++c) {
        float v = h[c];
        a0 += v * Wcomb[c * 2];
        a1 += v * Wcomb[c * 2 + 1];
    }
#pragma unroll 8
    for (int k = 0; k < OP_C; ++k) {
        float v = o[k];
        a0 += v * Wfc[(IN_C + k) * 2];
        a1 += v * Wfc[(IN_C + k) * 2 + 1];
    }
    out[(size_t)i * 2]     = a0;
    out[(size_t)i * 2 + 1] = a1;
}

// ------------------------------------------------------------- launch ----
extern "C" void kernel_launch(void* const* d_in, const int* in_sizes, int n_in,
                              void* d_out, int out_size, void* d_ws, size_t ws_size,
                              hipStream_t stream) {
    const float* x   = (const float*)d_in[0];
    const int*   ei  = (const int*)  d_in[1];
    const float* ea  = (const float*)d_in[2];
    const float* op  = (const float*)d_in[3];
    const float* W1  = (const float*)d_in[4];
    const float* b1  = (const float*)d_in[5];
    const float* W2  = (const float*)d_in[6];
    const float* b2  = (const float*)d_in[7];
    const float* Wp  = (const float*)d_in[8];
    const float* bp  = (const float*)d_in[9];
    const float* Wfc = (const float*)d_in[10];
    const float* bfc = (const float*)d_in[11];
    float* out = (float*)d_out;

    const int N = in_sizes[0] / IN_C;
    const int E = in_sizes[2];

    // workspace carve-up (256B aligned)
    char* ws = (char*)d_ws;
    size_t off = 0;
    auto take = [&](size_t bytes) { char* p = ws + off; off = (off + bytes + 255) & ~(size_t)255; return p; };
    float*    w_buf = (float*)   take((size_t)E * 4);
    float*    dinv  = (float*)   take((size_t)N * 4);          // deg -> dinv in place
    _Float16* W1t   = (_Float16*)take((size_t)IN_C * HID * 2);
    _Float16* W2t   = (_Float16*)take((size_t)HID * HID * 2);
    float*    Wcomb = (float*)   take((size_t)HID * OUT_C * 4);
    float*    bias2 = (float*)   take((size_t)OUT_C * 4);
    float*    bufA  = (float*)   take((size_t)N * HID * 4);
    float*    bufB  = (float*)   take((size_t)N * HID * 4);
    _Float16* Hf16  = (_Float16*)take((size_t)N * HID * 2);
    (void)ws_size; (void)n_in; (void)out_size;

    const int NB    = (N + 255) / 256;
    const int NHB   = (N * HID + 255) / 256;
    const int EB    = (E + 255) / 256;
    const int ECB   = ((E * 64) + 255) / 256;       // 64 channels per edge
    const int GEMMB = (N + 127) / 128;              // 128 rows per block (8 waves x 16)

    // degrees / weights / normalization
    k_init_deg   <<<NB, 256, 0, stream>>>(dinv, N);
    k_edge_w_deg <<<EB, 256, 0, stream>>>(ea, ei, w_buf, dinv, E);
    k_dinv       <<<NB, 256, 0, stream>>>(dinv, N);

    // weight prep
    k_transpose_f16<<<(IN_C * HID + 255) / 256, 256, 0, stream>>>(W1, W1t, IN_C, HID);
    k_transpose_f16<<<(HID * HID + 255) / 256, 256, 0, stream>>>(W2, W2t, HID, HID);
    k_prep_final   <<<1, 256, 0, stream>>>(Wp, bp, Wfc, bfc, Wcomb, bias2);

    // layer 1
    k_gemm_x_w1    <<<GEMMB, 256, 0, stream>>>(x, W1t, bufA, N);
    k_agg_self     <<<NHB, 256, 0, stream>>>(bufA, dinv, bufB, N);
    k_agg_edges    <<<ECB, 256, 0, stream>>>(bufA, ei, w_buf, dinv, bufB, E);
    k_bias_relu_f16<<<NHB, 256, 0, stream>>>(bufB, b1, Hf16, N);

    // layer 2
    k_gemm_h_w2    <<<GEMMB, 256, 0, stream>>>(Hf16, W2t, bufA, N);
    k_agg_self     <<<NHB, 256, 0, stream>>>(bufA, dinv, bufB, N);
    k_agg_edges    <<<ECB, 256, 0, stream>>>(bufA, ei, w_buf, dinv, bufB, E);
    k_bias_relu_f32<<<NHB, 256, 0, stream>>>(bufB, b2, bufA, N);

    // fused projection + concat + FC (Wp/Wfc algebraically folded)
    k_final<<<NB, 256, 0, stream>>>(bufA, op, Wcomb, Wfc, bias2, out, N);
}
